// MLAAttention_70523363000809
// MI455X (gfx1250) — matrix operand order repaired and checked by reading, not compiled, not run on hardware
//
#include <hip/hip_runtime.h>

// ---------------------------------------------------------------------------
// MLA attention for MI455X (gfx1250, wave32, WMMA).
// bf16 data path with fp32 WMMA accumulation; flash-style fused attention.
// GEMMs: double-buffered LDS pipeline using CDNA5 async global->LDS copies.
// ---------------------------------------------------------------------------

typedef __bf16 bf16_t;
typedef __attribute__((ext_vector_type(16))) bf16_t v16bf;
typedef __attribute__((ext_vector_type(8)))  bf16_t v8bf;
typedef __attribute__((ext_vector_type(8)))  float  v8f;
typedef unsigned short u16;

constexpr int B_   = 2;
constexpr int S_   = 2048;
constexpr int H_   = 16;
constexpr int HID_ = 2048;
constexpr int QR_  = 1536;
constexpr int KVR_ = 512;
constexpr int DQK_ = 192;   // 128 nope + 64 rope
constexpr int NT_  = B_ * S_;          // 4096 tokens
constexpr int NQB_ = H_ * DQK_;        // 3072
constexpr int NKVB_ = H_ * 256;        // 4096
constexpr int CKVW_ = KVR_ + 64;       // 576

static __device__ __forceinline__ u16 f2bf(float f) {
    unsigned int u = __float_as_uint(f);
    u += 0x7FFFu + ((u >> 16) & 1u);   // round-to-nearest-even
    return (u16)(u >> 16);
}

static __device__ __forceinline__ v16bf shuf16(v8bf lo, v8bf hi) {
    return __builtin_shufflevector(lo, hi, 0,1,2,3,4,5,6,7,8,9,10,11,12,13,14,15);
}

// ---------------------------------------------------------------------------
// CDNA5 async global -> LDS copy (ASYNCcnt-tracked), with safe fallback.
// Builtin signature (from hipcc diagnostic): (int4 AS1*, int4 AS3*, imm, imm).
// LDS byte offset = low 32 bits of the flat shared address (ISA 10.2 aperture
// truncation); global flat address == global address.
// ---------------------------------------------------------------------------
#if __has_builtin(__builtin_amdgcn_global_load_async_to_lds_b128)
#define ASYNC_LDS 1
typedef int v4i_t __attribute__((vector_size(16)));
typedef __attribute__((address_space(1))) v4i_t* as1_v4i_p;
typedef __attribute__((address_space(3))) v4i_t* as3_v4i_p;
#else
#define ASYNC_LDS 0
#endif

static __device__ __forceinline__ void cp16_g2s(const u16* g, u16* s) {
#if ASYNC_LDS
    __builtin_amdgcn_global_load_async_to_lds_b128(
        (as1_v4i_p)(size_t)g,
        (as3_v4i_p)(unsigned)(size_t)s,
        0, 0);
#else
    *(uint4*)s = *(const uint4*)g;
#endif
}

static __device__ __forceinline__ void async_wait0() {
#if ASYNC_LDS
#if __has_builtin(__builtin_amdgcn_s_wait_asynccnt)
    __builtin_amdgcn_s_wait_asynccnt(0);
#else
    asm volatile("s_wait_asynccnt 0x0" ::: "memory");
#endif
#endif
}

// ---------------------------------------------------------------------------
// fp32 -> bf16 convert (no transpose)
// ---------------------------------------------------------------------------
__global__ void conv_bf16_kernel(const float* __restrict__ x, u16* __restrict__ y, long n) {
    long i = (long)blockIdx.x * blockDim.x + threadIdx.x;
    if (i < n) y[i] = f2bf(x[i]);
}

// ---------------------------------------------------------------------------
// fp32 [K,N] -> bf16 transposed [N,K]
// ---------------------------------------------------------------------------
__global__ void convT_bf16_kernel(const float* __restrict__ w, u16* __restrict__ wt,
                                  int K, int N) {
    long i = (long)blockIdx.x * blockDim.x + threadIdx.x;
    long total = (long)K * N;
    if (i >= total) return;
    int k = (int)(i / N), n = (int)(i % N);
    wt[(size_t)n * K + k] = f2bf(w[i]);
}

// ---------------------------------------------------------------------------
// WMMA GEMM: C[M,N] fp32 = A[M,K] bf16 @ Bt[N,K] bf16 (B pre-transposed).
// Block tile 256x64, 8 waves each computing 64x32 (4x2 WMMA frags), K step 32.
// Double-buffered LDS; async global->LDS copies overlap the WMMA stream.
// ---------------------------------------------------------------------------
__global__ __launch_bounds__(256) void gemm_bf16_kernel(
    const u16* __restrict__ A, const u16* __restrict__ Bt, float* __restrict__ C,
    int M, int N, int K)
{
    __shared__ __align__(32) u16 As[2][256 * 32];   // 32 KB
    __shared__ __align__(32) u16 Bs[2][64 * 32];    //  8 KB

    const int tid  = threadIdx.x;
    const int lane = tid & 31, wv = tid >> 5;
    const int wm = wv >> 1;              // 0..3 : 64-row slab
    const int wn = wv & 1;               // 0..1 : 32-col slab
    const int hh = lane >> 4, ln = lane & 15;
    const int m0 = blockIdx.y * 256, n0 = blockIdx.x * 64;

    v8f acc[4][2];
    for (int i = 0; i < 4; ++i)
        for (int j = 0; j < 2; ++j)
            for (int r = 0; r < 8; ++r) acc[i][j][r] = 0.0f;

    // tile loaders: A 256x32 (32 elem/thread in two row-chunks), B 64x32 (8/thread)
    const int ar = tid >> 1, ac = (tid & 1) * 16;
    const int br = tid >> 2, bc = (tid & 3) * 8;

    auto load_tiles = [&](int buf, int k0) {
        const u16* ga0 = A + (size_t)(m0 + ar) * K + k0 + ac;
        u16* sa0 = As[buf] + ar * 32 + ac;
        cp16_g2s(ga0, sa0);
        cp16_g2s(ga0 + 8, sa0 + 8);
        const u16* ga1 = A + (size_t)(m0 + 128 + ar) * K + k0 + ac;
        u16* sa1 = As[buf] + (128 + ar) * 32 + ac;
        cp16_g2s(ga1, sa1);
        cp16_g2s(ga1 + 8, sa1 + 8);
        cp16_g2s(Bt + (size_t)(n0 + br) * K + k0 + bc, Bs[buf] + br * 32 + bc);
    };

    load_tiles(0, 0);
    int cur = 0;
    for (int k0 = 0; k0 < K; k0 += 32) {
        async_wait0();          // my async copies landed
        __syncthreads();        // everyone's copies landed / prior reads done
        if (k0 + 32 < K) load_tiles(cur ^ 1, k0 + 32);

        // A frag: lane 16h+m holds K = h*16+[0..15] of row m
        v16bf a[4], b[2];
        for (int fm = 0; fm < 4; ++fm) {
            const u16* p = As[cur] + (wm * 64 + fm * 16 + ln) * 32 + hh * 16;
            a[fm] = shuf16(*(const v8bf*)p, *(const v8bf*)(p + 8));
        }
        // B frag: lane 16h+n holds K = h*16+[0..15] of column n (contiguous in Bt)
        for (int fn = 0; fn < 2; ++fn)
            b[fn] = *(const v16bf*)(Bs[cur] + (wn * 32 + fn * 16 + ln) * 32 + hh * 16);

        for (int fm = 0; fm < 4; ++fm)
            for (int fn = 0; fn < 2; ++fn)
                acc[fm][fn] = __builtin_amdgcn_wmma_f32_16x16x32_bf16(
                    false, a[fm], false, b[fn], (short)0, acc[fm][fn], false, false);
        cur ^= 1;
    }

    // C frag: VGPR r, lane 16h+n -> M = r + 8h, N = n
    for (int fm = 0; fm < 4; ++fm)
        for (int fn = 0; fn < 2; ++fn)
            for (int r = 0; r < 8; ++r) {
                int row = m0 + wm * 64 + fm * 16 + 8 * hh + r;
                int col = n0 + wn * 32 + fn * 16 + ln;
                C[(size_t)row * N + col] = acc[fm][fn][r];
            }
}

// ---------------------------------------------------------------------------
// Row-wise RMSNorm (fp32 in, bf16 out):  y = x * rsqrt(mean(x^2)+eps) * g
// ---------------------------------------------------------------------------
__global__ __launch_bounds__(256) void rmsnorm_kernel(
    const float* __restrict__ x, int ldx, int cols,
    const float* __restrict__ g, u16* __restrict__ y, int ldy)
{
    __shared__ float red[256];
    const int row = blockIdx.x, tid = threadIdx.x;
    const float* xr = x + (size_t)row * ldx;
    float s = 0.0f;
    for (int c = tid; c < cols; c += 256) { float v = xr[c]; s += v * v; }
    red[tid] = s; __syncthreads();
    for (int st = 128; st > 0; st >>= 1) {
        if (tid < st) red[tid] += red[tid + st];
        __syncthreads();
    }
    const float rs = rsqrtf(red[0] / (float)cols + 1e-6f);
    u16* yr = y + (size_t)row * ldy;
    for (int c = tid; c < cols; c += 256) yr[c] = f2bf(xr[c] * rs * g[c]);
}

// ---------------------------------------------------------------------------
// Build qf[B,H,S,192] bf16: copy q_pass, interleave+RoPE q_rot.
// interleave: y[2i]=x[i], y[2i+1]=x[32+i]; rope: y*c + rotate_half(y)*s
// ---------------------------------------------------------------------------
__global__ void build_qf_kernel(const float* __restrict__ q, const float* __restrict__ cs,
                                const float* __restrict__ sn, u16* __restrict__ qf)
{
    const int s = blockIdx.x, h = blockIdx.y, b = blockIdx.z, d = threadIdx.x;
    const size_t tok = (size_t)b * S_ + s;
    const float* qr = q + tok * NQB_ + (size_t)h * DQK_;
    u16* o = qf + (((size_t)b * H_ + h) * S_ + s) * DQK_;
    if (d < 128) { o[d] = f2bf(qr[d]); return; }
    const int j = d - 128;
    const float* x = qr + 128;
    auto yv = [&](int jj) { return (jj & 1) ? x[32 + (jj >> 1)] : x[jj >> 1]; };
    const float c = cs[tok * 64 + j], si = sn[tok * 64 + j];
    const float rh = (j < 32) ? -yv(j + 32) : yv(j - 32);
    o[d] = f2bf(yv(j) * c + rh * si);
}

// ---------------------------------------------------------------------------
// Build kf[B,H,S,192] bf16 and vT[B,H,128,S] bf16 (V transposed for WMMA-B).
// ---------------------------------------------------------------------------
__global__ void build_kfv_kernel(const float* __restrict__ kv, const float* __restrict__ ckv,
                                 const float* __restrict__ cs, const float* __restrict__ sn,
                                 u16* __restrict__ kf, u16* __restrict__ vT)
{
    const int s = blockIdx.x, h = blockIdx.y, b = blockIdx.z, d = threadIdx.x;
    const size_t tok = (size_t)b * S_ + s;
    const float* kvr = kv + tok * NKVB_ + (size_t)h * 256;
    u16* o = kf + (((size_t)b * H_ + h) * S_ + s) * DQK_;
    if (d < 128) {
        o[d] = f2bf(kvr[d]);                                       // k_nope
        vT[(((size_t)b * H_ + h) * 128 + d) * S_ + s] = f2bf(kvr[128 + d]);
        return;
    }
    const int j = d - 128;                                          // roped k (bcast over h)
    const float* x = ckv + tok * CKVW_ + KVR_;
    auto yv = [&](int jj) { return (jj & 1) ? x[32 + (jj >> 1)] : x[jj >> 1]; };
    const float c = cs[tok * 64 + j], si = sn[tok * 64 + j];
    const float rh = (j < 32) ? -yv(j + 32) : yv(j - 32);
    o[d] = f2bf(yv(j) * c + rh * si);
}

// ---------------------------------------------------------------------------
// Flash attention. Block = 4 waves, one (b,h,64-query) tile per block.
// Each wave: 16 query rows; streams causal 64-key tiles.
//   scores: 4 ntiles x 6 kstep WMMAs ; PV: 2 kstep x 8 vtile WMMAs
// P re-laid out C-frag -> A-frag through wave-private LDS (in-order DS ops).
// ---------------------------------------------------------------------------
__global__ __launch_bounds__(128) void mla_flash_kernel(
    const u16* __restrict__ qf, const u16* __restrict__ kf,
    const u16* __restrict__ vT, u16* __restrict__ obf)
{
    __shared__ __align__(32) u16 Ps[4][16][64];
    const int tid = threadIdx.x, lane = tid & 31, w = tid >> 5;
    const int hh = lane >> 4, ln = lane & 15;
    const int qb = blockIdx.x, h = blockIdx.y, b = blockIdx.z;
    const size_t bh = (size_t)b * H_ + h;
    const int q0 = qb * 64 + w * 16;

    // Q fragments: 16 rows x 192 (6 k-steps), direct global loads in A-frag layout
    v16bf qa[6];
    {
        const u16* qrow = qf + (bh * S_ + q0) * DQK_ + (size_t)ln * DQK_;
        for (int j = 0; j < 6; ++j) {
            const u16* p = qrow + j * 32 + hh * 16;
            qa[j] = shuf16(*(const v8bf*)p, *(const v8bf*)(p + 8));
        }
    }

    float mrow[8], lrow[8];
    v8f vacc[8];
    for (int r = 0; r < 8; ++r) { mrow[r] = -3.0e38f; lrow[r] = 0.0f; }
    for (int vt = 0; vt < 8; ++vt)
        for (int r = 0; r < 8; ++r) vacc[vt][r] = 0.0f;

    const float scale = 0.07216878364870323f;   // 1/sqrt(192)

    for (int jt = 0; jt <= qb; ++jt) {
        const int k0 = jt * 64;

        // ---- S = Q @ K^T  (kf row-major is exactly WMMA-B layout source) ----
        v8f sacc[4];
        for (int nt = 0; nt < 4; ++nt)
            for (int r = 0; r < 8; ++r) sacc[nt][r] = 0.0f;
        for (int nt = 0; nt < 4; ++nt) {
            const u16* krow = kf + (bh * S_ + k0 + nt * 16 + ln) * DQK_;
            for (int ks = 0; ks < 6; ++ks) {
                v16bf bk = *(const v16bf*)(krow + ks * 32 + hh * 16);
                sacc[nt] = __builtin_amdgcn_wmma_f32_16x16x32_bf16(
                    false, qa[ks], false, bk, (short)0, sacc[nt], false, false);
            }
        }

        // ---- online softmax over this key tile ----
        for (int r = 0; r < 8; ++r) {
            const int qg = q0 + 8 * hh + r;
            float mx = mrow[r];
            float sc[4];
            for (int nt = 0; nt < 4; ++nt) {
                const int kk = k0 + nt * 16 + ln;
                float sv = sacc[nt][r] * scale;
                sv = (kk > qg) ? -3.0e38f : sv;      // causal mask
                sc[nt] = sv;
                mx = fmaxf(mx, sv);
            }
            for (int off = 1; off < 16; off <<= 1)
                mx = fmaxf(mx, __shfl_xor(mx, off, 32));
            const float corr = __expf(mrow[r] - mx);
            mrow[r] = mx;
            for (int vt = 0; vt < 8; ++vt) vacc[vt][r] *= corr;
            float ps = 0.0f;
            for (int nt = 0; nt < 4; ++nt) {
                const float p = __expf(sc[nt] - mx);
                ps += p;
                Ps[w][8 * hh + r][nt * 16 + ln] = f2bf(p);
            }
            for (int off = 1; off < 16; off <<= 1)
                ps += __shfl_xor(ps, off, 32);
            lrow[r] = lrow[r] * corr + ps;
        }

        // ---- O += P @ V  (vT gives contiguous K for WMMA-B fragments) ----
        for (int ks2 = 0; ks2 < 2; ++ks2) {
            const u16* pr = &Ps[w][ln][ks2 * 32 + hh * 16];
            v16bf ap = shuf16(*(const v8bf*)pr, *(const v8bf*)(pr + 8));
            for (int vt = 0; vt < 8; ++vt) {
                const u16* vrow = vT + (bh * 128 + vt * 16 + ln) * (size_t)S_
                                     + k0 + ks2 * 32 + hh * 16;
                v16bf bv = *(const v16bf*)vrow;
                vacc[vt] = __builtin_amdgcn_wmma_f32_16x16x32_bf16(
                    false, ap, false, bv, (short)0, vacc[vt], false, false);
            }
        }
    }

    // epilogue: normalize and store as [B,S,H*128] bf16 (head transpose fused)
    for (int vt = 0; vt < 8; ++vt)
        for (int r = 0; r < 8; ++r) {
            const int row = q0 + 8 * hh + r;
            const float o = vacc[vt][r] / lrow[r];
            obf[((size_t)b * S_ + row) * (H_ * 128) + h * 128 + vt * 16 + ln] = f2bf(o);
        }
}

// ---------------------------------------------------------------------------
// Host-side launch
// ---------------------------------------------------------------------------
extern "C" void kernel_launch(void* const* d_in, const int* in_sizes, int n_in,
                              void* d_out, int out_size, void* d_ws, size_t ws_size,
                              hipStream_t stream)
{
    (void)in_sizes; (void)n_in; (void)out_size; (void)ws_size;

    const float* hs    = (const float*)d_in[0];
    const float* w_qa  = (const float*)d_in[1];
    const float* g_qa  = (const float*)d_in[2];
    const float* w_qb  = (const float*)d_in[3];
    const float* w_kva = (const float*)d_in[4];
    const float* g_kva = (const float*)d_in[5];
    const float* w_kvb = (const float*)d_in[6];
    const float* w_o   = (const float*)d_in[7];
    const float* cs    = (const float*)d_in[8];
    const float* sn    = (const float*)d_in[9];
    float* out = (float*)d_out;

    // workspace carve-out (256B aligned regions)
    size_t off = 0;
    auto alloc = [&](size_t bytes) -> void* {
        void* p = (char*)d_ws + off;
        off += (bytes + 255) & ~(size_t)255;
        return p;
    };
    u16*   hs_bf  = (u16*)  alloc((size_t)NT_ * HID_ * 2);
    u16*   wbuf   = (u16*)  alloc((size_t)3072 * 1536 * 2);    // max weight (transposed)
    float* q_a    = (float*)alloc((size_t)NT_ * QR_ * 4);
    u16*   qa_n   = (u16*)  alloc((size_t)NT_ * QR_ * 2);
    float* qfull  = (float*)alloc((size_t)NT_ * NQB_ * 4);
    float* ckv    = (float*)alloc((size_t)NT_ * CKVW_ * 4);
    u16*   ckv_n  = (u16*)  alloc((size_t)NT_ * KVR_ * 2);
    float* kv     = (float*)alloc((size_t)NT_ * NKVB_ * 4);
    u16*   qfb    = (u16*)  alloc((size_t)NT_ * H_ * DQK_ * 2);
    u16*   kfb    = (u16*)  alloc((size_t)NT_ * H_ * DQK_ * 2);
    u16*   vTb    = (u16*)  alloc((size_t)NT_ * H_ * 128 * 2);
    u16*   obf    = (u16*)  alloc((size_t)NT_ * HID_ * 2);

    auto cdiv = [](long a, long b) { return (unsigned)((a + b - 1) / b); };

    // hidden_states -> bf16
    conv_bf16_kernel<<<cdiv((long)NT_ * HID_, 256), 256, 0, stream>>>(hs, hs_bf, (long)NT_ * HID_);

    // q_a = hs @ w_qa ; rmsnorm
    convT_bf16_kernel<<<cdiv((long)HID_ * QR_, 256), 256, 0, stream>>>(w_qa, wbuf, HID_, QR_);
    gemm_bf16_kernel<<<dim3(QR_ / 64, NT_ / 256), 256, 0, stream>>>(hs_bf, wbuf, q_a, NT_, QR_, HID_);
    rmsnorm_kernel<<<NT_, 256, 0, stream>>>(q_a, QR_, QR_, g_qa, qa_n, QR_);

    // q = qa_n @ w_qb
    convT_bf16_kernel<<<cdiv((long)QR_ * NQB_, 256), 256, 0, stream>>>(w_qb, wbuf, QR_, NQB_);
    gemm_bf16_kernel<<<dim3(NQB_ / 64, NT_ / 256), 256, 0, stream>>>(qa_n, wbuf, qfull, NT_, NQB_, QR_);

    // ckv = hs @ w_kva ; rmsnorm of first 512 cols
    convT_bf16_kernel<<<cdiv((long)HID_ * CKVW_, 256), 256, 0, stream>>>(w_kva, wbuf, HID_, CKVW_);
    gemm_bf16_kernel<<<dim3(CKVW_ / 64, NT_ / 256), 256, 0, stream>>>(hs_bf, wbuf, ckv, NT_, CKVW_, HID_);
    rmsnorm_kernel<<<NT_, 256, 0, stream>>>(ckv, CKVW_, KVR_, g_kva, ckv_n, KVR_);

    // kv = ckv_n @ w_kvb
    convT_bf16_kernel<<<cdiv((long)KVR_ * NKVB_, 256), 256, 0, stream>>>(w_kvb, wbuf, KVR_, NKVB_);
    gemm_bf16_kernel<<<dim3(NKVB_ / 64, NT_ / 256), 256, 0, stream>>>(ckv_n, wbuf, kv, NT_, NKVB_, KVR_);

    // RoPE + layout for attention
    build_qf_kernel<<<dim3(S_, H_, B_), 192, 0, stream>>>(qfull, cs, sn, qfb);
    build_kfv_kernel<<<dim3(S_, H_, B_), 192, 0, stream>>>(kv, ckv, cs, sn, kfb, vTb);

    // fused causal attention
    mla_flash_kernel<<<dim3(S_ / 64, H_, B_), 128, 0, stream>>>(qfb, kfb, vTb, obf);

    // out = attn_out @ w_o
    convT_bf16_kernel<<<cdiv((long)HID_ * HID_, 256), 256, 0, stream>>>(w_o, wbuf, HID_, HID_);
    gemm_bf16_kernel<<<dim3(HID_ / 64, NT_ / 256), 256, 0, stream>>>(obf, wbuf, out, NT_, HID_, HID_);
}
